// SelfAttention_21225728377292
// MI455X (gfx1250) — matrix operand order repaired and checked by reading, not compiled, not run on hardware
//
#include <hip/hip_runtime.h>
#include <hip/hip_bf16.h>
#include <math.h>

// ---------------------------------------------------------------------------
// MI455X (gfx1250) implementation. wave32, WMMA f32<-f16 16x16x32.
// All GEMM-like math runs through v_wmma_f32_16x16x32_f16 with f32 accum.
// Attention is flash-style (streaming softmax) so the 4096x4096 score matrix
// is never materialized (saves ~1.3 GB HBM traffic per attention block).
// Round-2 changes: K-chunk 64 (8 WMMAs per barrier pair in GEMM, 4 in conv),
// float4/v4h vectorized flash tile fills, global_prefetch for next K-chunk.
// ---------------------------------------------------------------------------

typedef __attribute__((ext_vector_type(16))) _Float16 v16h;
typedef __attribute__((ext_vector_type(8)))  _Float16 v8h;
typedef __attribute__((ext_vector_type(4)))  _Float16 v4h;
typedef __attribute__((ext_vector_type(8)))  float    v8f;

static __device__ __forceinline__ v8f wmma_f16(v16h a, v16h b, v8f c) {
  // 8 args: (neg_a, A, neg_b, B, c_mod, C, reuse_a, reuse_b)
  return __builtin_amdgcn_wmma_f32_16x16x32_f16(false, a, false, b, (short)0, c,
                                                false, false);
}

// A fragment (16x32 f16) from LDS tile stored [M][K] row-major.
// Lanes 0-15: row=lane, K 0..7 & 16..23 ; lanes 16-31: row=lane-16, K 8..15 & 24..31.
static __device__ __forceinline__ v16h frag_a_lds(const _Float16* t, int ld,
                                                  int row0, int k0) {
  int lane = threadIdx.x & 31;
  const _Float16* p = t + (row0 + (lane & 15)) * ld + k0 + ((lane >> 4) << 3);
  v8h lo = *(const v8h*)p;
  v8h hi = *(const v8h*)(p + 16);
  v16h r;
#pragma unroll
  for (int i = 0; i < 8; ++i) { r[i] = lo[i]; r[i + 8] = hi[i]; }
  return r;
}

// B fragment (32x16 f16) from LDS tile stored transposed [N][K].
// Lanes 0-15: col=lane, K 0..7 & 16..23 ; lanes 16-31: col=lane-16, K 8..15 & 24..31.
static __device__ __forceinline__ v16h frag_b_lds(const _Float16* t, int ld,
                                                  int col0, int k0) {
  int lane = threadIdx.x & 31;
  const _Float16* p = t + (col0 + (lane & 15)) * ld + k0 + ((lane >> 4) << 3);
  v8h lo = *(const v8h*)p;
  v8h hi = *(const v8h*)(p + 16);
  v16h r;
#pragma unroll
  for (int i = 0; i < 8; ++i) { r[i] = lo[i]; r[i + 8] = hi[i]; }
  return r;
}

static __device__ __forceinline__ float sigmoidf_(float x) {
  return 1.0f / (1.0f + __expf(-x));
}

// ---------------------------------------------------------------------------
// Generic C[M,N] = W[M,K] * X[K,N] + bias[M], optional ReLU. Batched over z.
// Block tile 64(M) x 128(N), K-chunk 64 (K must be a multiple of 64).
// 8 waves, each wave a 32x32 tile -> 8 WMMAs per barrier pair.
// ---------------------------------------------------------------------------
template <int ACT>
__global__ __launch_bounds__(256) void gemm_wx(
    const float* __restrict__ W, const float* __restrict__ X,
    const float* __restrict__ bias, float* __restrict__ C, int M, int K, int N,
    long long xBatch, long long cBatch) {
  __shared__ _Float16 As[64 * 72];   // [M][K] ld=72 (pad)
  __shared__ _Float16 Bs[128 * 72];  // [N][K] ld=72
  const int tid = threadIdx.x, lane = tid & 31, wid = tid >> 5;
  const int m0 = blockIdx.y * 64, n0 = blockIdx.x * 128;
  const float* Xb = X + (long long)blockIdx.z * xBatch;
  float* Cb = C + (long long)blockIdx.z * cBatch;
  const int wm = wid & 1, wn = wid >> 1;  // 2x4 wave grid
  const int am = tid >> 2, ak = (tid & 3) * 16;       // A fill coords
  const int bk = tid >> 2, bn = (tid & 3) * 32;       // B fill coords
  v8f acc[2][2] = {};
  for (int k0 = 0; k0 < K; k0 += 64) {
    __syncthreads();
    {  // A: 64x64 = 4096 elems, 16/thread, coalesced along K
      const float* src = W + (long long)(m0 + am) * K + k0 + ak;
#pragma unroll
      for (int i = 0; i < 16; ++i) As[am * 72 + ak + i] = (_Float16)src[i];
    }
    {  // B: 64x128 = 8192 elems, 32/thread, coalesced along N, transposed store
      const float* src = Xb + (long long)(k0 + bk) * N + n0 + bn;
#pragma unroll
      for (int i = 0; i < 32; ++i) Bs[(bn + i) * 72 + bk] = (_Float16)src[i];
    }
    if (k0 + 64 < K) {  // prefetch next K-chunk (global_prefetch_b8)
      __builtin_prefetch(W + (long long)(m0 + am) * K + k0 + 64 + ak, 0, 0);
      __builtin_prefetch(Xb + (long long)(k0 + 64 + bk) * N + n0 + bn, 0, 0);
    }
    __syncthreads();
#pragma unroll
    for (int ks = 0; ks < 2; ++ks) {
      v16h a0 = frag_a_lds(As, 72, wm * 32, ks * 32);
      v16h a1 = frag_a_lds(As, 72, wm * 32 + 16, ks * 32);
      v16h b0 = frag_b_lds(Bs, 72, wn * 32, ks * 32);
      v16h b1 = frag_b_lds(Bs, 72, wn * 32 + 16, ks * 32);
      acc[0][0] = wmma_f16(a0, b0, acc[0][0]);
      acc[0][1] = wmma_f16(a0, b1, acc[0][1]);
      acc[1][0] = wmma_f16(a1, b0, acc[1][0]);
      acc[1][1] = wmma_f16(a1, b1, acc[1][1]);
    }
  }
  const int hi8 = (lane >> 4) << 3, l15 = lane & 15;
#pragma unroll
  for (int tm = 0; tm < 2; ++tm)
#pragma unroll
    for (int tn = 0; tn < 2; ++tn)
#pragma unroll
      for (int e = 0; e < 8; ++e) {
        int row = m0 + wm * 32 + tm * 16 + e + hi8;
        int col = n0 + wn * 32 + tn * 16 + l15;
        float v = acc[tm][tn][e] + bias[row];
        if (ACT == 1) v = fmaxf(v, 0.0f);
        Cb[(long long)row * N + col] = v;
      }
}

// ---------------------------------------------------------------------------
// 7x7 conv (pad 3) as implicit GEMM + BN (+opt ReLU). In/Out are (Cin,64,64)
// / (O,64,64) per batch. One block = 64 out-channels x one image row (64 px).
// C-chunk 64 (Cin multiple of 64): 4 WMMAs per wave per barrier pair.
// ---------------------------------------------------------------------------
template <int ACT>
__global__ __launch_bounds__(256) void conv7_bn(
    const float* __restrict__ Wc, const float* __restrict__ bias,
    const float* __restrict__ g, const float* __restrict__ bb,
    const float* __restrict__ mm, const float* __restrict__ vv,
    const float* __restrict__ In, float* __restrict__ Out, int O, int Cin,
    long long inBatch, long long outBatch) {
  __shared__ _Float16 As[64 * 72];  // [o][c]
  __shared__ _Float16 Bs[64 * 72];  // [x][c]
  const int tid = threadIdx.x, lane = tid & 31, wid = tid >> 5;
  const int y = blockIdx.x, o0 = blockIdx.y * 64;
  const float* Ib = In + (long long)blockIdx.z * inBatch;
  float* Ob = Out + (long long)blockIdx.z * outBatch;
  const int wm = wid >> 1, wn = wid & 1;  // 4x2 wave grid (16x32 per wave)
  const int ao = tid >> 2, ac = (tid & 3) * 16;  // weight fill coords
  const int bc = tid >> 2, bx = (tid & 3) * 16;  // input fill coords
  v8f acc[2] = {};
  for (int ky = 0; ky < 7; ++ky) {
    int iy = y + ky - 3;
    bool rowOk = (iy >= 0) && (iy < 64);
    for (int kx = 0; kx < 7; ++kx) {
      for (int c0 = 0; c0 < Cin; c0 += 64) {
        __syncthreads();
        {  // weights: stride-49 gathers, 16/thread
          const float* src =
              Wc + ((long long)(o0 + ao) * Cin + c0 + ac) * 49 + ky * 7 + kx;
#pragma unroll
          for (int i = 0; i < 16; ++i)
            As[ao * 72 + ac + i] = (_Float16)src[(long long)i * 49];
          if (c0 + 64 < Cin) __builtin_prefetch(src + 64 * 49, 0, 0);
        }
        {  // shifted input row with zero padding, 16/thread
#pragma unroll
          for (int i = 0; i < 16; ++i) {
            int x = bx + i, ix = x + kx - 3;
            float vval = 0.0f;
            if (rowOk && ix >= 0 && ix < 64)
              vval = Ib[(long long)(c0 + bc) * 4096 + iy * 64 + ix];
            Bs[x * 72 + bc] = (_Float16)vval;
          }
          if (c0 + 64 < Cin && rowOk)
            __builtin_prefetch(Ib + (long long)(c0 + 64 + bc) * 4096 + iy * 64,
                               0, 0);
        }
        __syncthreads();
#pragma unroll
        for (int ks = 0; ks < 2; ++ks) {
          v16h a = frag_a_lds(As, 72, wm * 16, ks * 32);
          v16h b0 = frag_b_lds(Bs, 72, wn * 32, ks * 32);
          v16h b1 = frag_b_lds(Bs, 72, wn * 32 + 16, ks * 32);
          acc[0] = wmma_f16(a, b0, acc[0]);
          acc[1] = wmma_f16(a, b1, acc[1]);
        }
      }
    }
  }
  const int hi8 = (lane >> 4) << 3, l15 = lane & 15;
#pragma unroll
  for (int tn = 0; tn < 2; ++tn)
#pragma unroll
    for (int e = 0; e < 8; ++e) {
      int o = o0 + wm * 16 + e + hi8;
      int x = wn * 32 + tn * 16 + l15;
      float val = acc[tn][e] + bias[o];
      float inv = g[o] * rsqrtf(vv[o] + 1e-5f);
      val = (val - mm[o]) * inv + bb[o];
      if (ACT == 1) val = fmaxf(val, 0.0f);
      Ob[(long long)o * 4096 + y * 64 + x] = val;
    }
}

// ---------------------------------------------------------------------------
// Flash attention: O[n,c] = softmax_m(Q[n,:].K[m,:]*scale) @ V[m,:].
// Row block 64, col block 64, D_V=512. 8 waves; wave w owns cols [64w,64w+64)
// => 16 f32 16x16 accumulators (128 VGPRs) per wave. Online softmax in LDS.
// REQUIRES: qsn==1, ksn==1, vsm==1 (n/m contiguous), i.e. (D,N)-major inputs.
// Optional epilogue: O = gamma*O + Res (first attention residual).
// ---------------------------------------------------------------------------
__global__ __launch_bounds__(256) void flash_attn(
    const float* __restrict__ Qg, long long qsd, long long qB,
    const float* __restrict__ Kg, long long ksd, long long kB,
    const float* __restrict__ Vg, long long vsc, long long vB,
    float* __restrict__ Og, long long osn, long long osc, long long oB, int N,
    int D, int Dlog, float scale, const float* __restrict__ gammaPtr,
    const float* __restrict__ Res, long long resB) {
  extern __shared__ char smem[];
  const int Dpad = D + 8;
  _Float16* Qs = (_Float16*)smem;        // [64][Dpad]  (A layout, pre-scaled)
  _Float16* Ks = Qs + 64 * Dpad;         // [64][72]    (B layout [j][d])
  _Float16* Vs = Ks + 64 * 72;           // [512][72]   (B layout [c][j])
  _Float16* Ps = Vs + 512 * 72;          // [64][72]    (A layout [i][j])
  float* Ss = (float*)(Ps + 64 * 72);    // [64][64]
  float* mrow = Ss + 64 * 64;
  float* lrow = mrow + 64;
  float* arow = lrow + 64;

  const int tid = threadIdx.x, lane = tid & 31, wid = tid >> 5;
  const int hi8 = (lane >> 4) << 3, l15 = lane & 15;
  const int i0 = blockIdx.x * 64;
  const float* Q = Qg + (long long)blockIdx.z * qB;
  const float* K = Kg + (long long)blockIdx.z * kB;
  const float* V = Vg + (long long)blockIdx.z * vB;
  float* Opt = Og + (long long)blockIdx.z * oB;

  // Q tile fill: float4 along n (contiguous), transposed scatter into [i][d].
  for (int q = tid; q < 16 * D; q += 256) {
    int i4 = (q >> Dlog) << 2, d = q & (D - 1);
    float4 qv = *(const float4*)(Q + (long long)(i0 + i4) + (long long)d * qsd);
    Qs[(i4 + 0) * Dpad + d] = (_Float16)(qv.x * scale);
    Qs[(i4 + 1) * Dpad + d] = (_Float16)(qv.y * scale);
    Qs[(i4 + 2) * Dpad + d] = (_Float16)(qv.z * scale);
    Qs[(i4 + 3) * Dpad + d] = (_Float16)(qv.w * scale);
  }
  if (tid < 64) { mrow[tid] = -1e30f; lrow[tid] = 0.0f; }
  v8f acc[4][4] = {};
  __syncthreads();

  const int sm = wid >> 1;           // wave's S row tile (0..3)
  const int snb = (wid & 1) * 2;     // wave's S col tile pair base
  for (int j0 = 0; j0 < N; j0 += 64) {
    // ---- S = Q.K^T (64x64), D chunked by 64 ----
    v8f sacc[2] = {};
    for (int d0 = 0; d0 < D; d0 += 64) {
      // K tile fill: float4 along j, transposed scatter into [j][d].
      for (int q = tid; q < 1024; q += 256) {
        int dd = q & 63, j4 = (q >> 6) << 2;
        float4 kv = *(const float4*)(K + (long long)(j0 + j4) +
                                     (long long)(d0 + dd) * ksd);
        Ks[(j4 + 0) * 72 + dd] = (_Float16)kv.x;
        Ks[(j4 + 1) * 72 + dd] = (_Float16)kv.y;
        Ks[(j4 + 2) * 72 + dd] = (_Float16)kv.z;
        Ks[(j4 + 3) * 72 + dd] = (_Float16)kv.w;
      }
      __syncthreads();
#pragma unroll
      for (int ks = 0; ks < 2; ++ks) {
        v16h a = frag_a_lds(Qs, Dpad, sm * 16, d0 + ks * 32);
        v16h b0 = frag_b_lds(Ks, 72, (snb + 0) * 16, ks * 32);
        v16h b1 = frag_b_lds(Ks, 72, (snb + 1) * 16, ks * 32);
        sacc[0] = wmma_f16(a, b0, sacc[0]);
        sacc[1] = wmma_f16(a, b1, sacc[1]);
      }
      __syncthreads();
    }
#pragma unroll
    for (int t = 0; t < 2; ++t)
#pragma unroll
      for (int e = 0; e < 8; ++e)
        Ss[(sm * 16 + e + hi8) * 64 + (snb + t) * 16 + l15] = sacc[t][e];
    __syncthreads();
    // ---- online softmax (one row per thread, threads 0..63) ----
    if (tid < 64) {
      float mo = mrow[tid], mx = mo;
      for (int c = 0; c < 64; ++c) mx = fmaxf(mx, Ss[tid * 64 + c]);
      float alpha = __expf(mo - mx);
      float s = 0.0f;
      for (int c = 0; c < 64; ++c) {
        float p = __expf(Ss[tid * 64 + c] - mx);
        s += p;
        Ps[tid * 72 + c] = (_Float16)p;
      }
      lrow[tid] = lrow[tid] * alpha + s;
      mrow[tid] = mx;
      arow[tid] = alpha;
    }
    __syncthreads();
    // ---- V tile fill (rows contiguous in j): float4 loads + v4h stores ----
    for (int q = tid; q < 8192; q += 256) {
      int c = q >> 4, j4 = (q & 15) << 2;
      float4 vv = *(const float4*)(V + (long long)(j0 + j4) +
                                   (long long)c * vsc);
      v4h h = {(_Float16)vv.x, (_Float16)vv.y, (_Float16)vv.z, (_Float16)vv.w};
      *(v4h*)(Vs + c * 72 + j4) = h;
    }
    // rescale accumulators while V-tile loads are in flight
#pragma unroll
    for (int tm = 0; tm < 4; ++tm)
#pragma unroll
      for (int e = 0; e < 8; ++e) {
        float al = arow[tm * 16 + e + hi8];
#pragma unroll
        for (int tn = 0; tn < 4; ++tn) acc[tm][tn][e] *= al;
      }
    __syncthreads();
    // ---- O += P.V (wave cols = wid*64 .. +64): 32 WMMAs/wave ----
#pragma unroll
    for (int ks = 0; ks < 2; ++ks) {
      v16h bf[4];
#pragma unroll
      for (int tn = 0; tn < 4; ++tn)
        bf[tn] = frag_b_lds(Vs, 72, wid * 64 + tn * 16, ks * 32);
#pragma unroll
      for (int tm = 0; tm < 4; ++tm) {
        v16h a = frag_a_lds(Ps, 72, tm * 16, ks * 32);
#pragma unroll
        for (int tn = 0; tn < 4; ++tn) acc[tm][tn] = wmma_f16(a, bf[tn], acc[tm][tn]);
      }
    }
    __syncthreads();
  }
  float gam = gammaPtr ? gammaPtr[0] : 1.0f;
#pragma unroll
  for (int tm = 0; tm < 4; ++tm)
#pragma unroll
    for (int e = 0; e < 8; ++e) {
      int rl = tm * 16 + e + hi8;
      float invl = 1.0f / lrow[rl];
      long long n = i0 + rl;
#pragma unroll
      for (int tn = 0; tn < 4; ++tn) {
        long long c = wid * 64 + tn * 16 + l15;
        long long off = n * osn + c * osc;
        float val = acc[tm][tn][e] * invl;
        if (Res) val = gam * val + Res[(long long)blockIdx.z * resB + off];
        Opt[off] = val;
      }
    }
}

// ---------------------------------------------------------------------------
// Elementwise / reduction kernels
// ---------------------------------------------------------------------------
__global__ __launch_bounds__(256) void ew_mul(const float* __restrict__ A,
                                              const float* __restrict__ B,
                                              float* __restrict__ C,
                                              long long n) {
  long long i = (long long)blockIdx.x * 256 + threadIdx.x;
  if (i < n) C[i] = A[i] * B[i];
}

// simam fused with sigmoid spatial gate: v = G*sigmoid(Sp); per (b,c) plane
// of 4096 px compute mean/var in one pass (values kept in registers), then
// out = v * sigmoid((v-mu)^2/(4*(var_sum/n + eps)) + 0.5).
__global__ __launch_bounds__(256) void simam_gate(const float* __restrict__ G,
                                                  const float* __restrict__ Sp,
                                                  float* __restrict__ Out) {
  __shared__ float r1[256], r2[256];
  long long base = (long long)blockIdx.x * 4096;
  float vals[16];
  float s = 0.f, s2 = 0.f;
#pragma unroll
  for (int t = 0; t < 16; ++t) {
    int idx = threadIdx.x + t * 256;
    float v = G[base + idx] * sigmoidf_(Sp[base + idx]);
    vals[t] = v;
    s += v;
    s2 += v * v;
  }
  r1[threadIdx.x] = s;
  r2[threadIdx.x] = s2;
  __syncthreads();
  for (int off = 128; off > 0; off >>= 1) {
    if (threadIdx.x < off) {
      r1[threadIdx.x] += r1[threadIdx.x + off];
      r2[threadIdx.x] += r2[threadIdx.x + off];
    }
    __syncthreads();
  }
  float S = r1[0], S2 = r2[0];
  float mean = S * (1.0f / 4096.0f);
  float d2sum = S2 - S * mean;
  float rden = 1.0f / (4.0f * (d2sum * (1.0f / 4095.0f) + 1e-4f));
#pragma unroll
  for (int t = 0; t < 16; ++t) {
    int idx = threadIdx.x + t * 256;
    float v = vals[t];
    float d = v - mean;
    Out[base + idx] = v * sigmoidf_(d * d * rden + 0.5f);
  }
}

// out[b,c,i,j] = P[b][c*4096 + j*64 + i]  (the final permute(0,3,2,1))
__global__ __launch_bounds__(256) void final_permute(const float* __restrict__ P,
                                                     float* __restrict__ Out) {
  long long idx = (long long)blockIdx.x * 256 + threadIdx.x;
  int j = (int)(idx & 63);
  int i = (int)((idx >> 6) & 63);
  long long bc = idx >> 12;
  Out[idx] = P[bc * 4096 + (long long)j * 64 + i];
}

// ---------------------------------------------------------------------------
static size_t flash_smem_bytes(int D) {
  size_t halves = (size_t)64 * (D + 8) + 64 * 72 + 512 * 72 + 64 * 72;
  return halves * 2 + (size_t)64 * 64 * 4 + 3 * 64 * 4;
}

extern "C" void kernel_launch(void* const* d_in, const int* in_sizes, int n_in,
                              void* d_out, int out_size, void* d_ws,
                              size_t ws_size, hipStream_t stream) {
  (void)in_sizes; (void)n_in; (void)out_size; (void)ws_size;
  const float* x       = (const float*)d_in[0];
  const float* Wq      = (const float*)d_in[1];
  const float* bq      = (const float*)d_in[2];
  const float* Wk      = (const float*)d_in[3];
  const float* bk      = (const float*)d_in[4];
  const float* Wv      = (const float*)d_in[5];
  const float* bv      = (const float*)d_in[6];
  const float* gamma   = (const float*)d_in[7];
  const float* W_qkv   = (const float*)d_in[8];
  const float* b_qkv   = (const float*)d_in[9];
  const float* W_proj  = (const float*)d_in[10];
  const float* b_proj  = (const float*)d_in[11];
  const float* gam_W1  = (const float*)d_in[12];
  const float* gam_b1  = (const float*)d_in[13];
  const float* gam_W2  = (const float*)d_in[14];
  const float* gam_b2  = (const float*)d_in[15];
  const float* gam_Wc1 = (const float*)d_in[16];
  const float* gam_bc1 = (const float*)d_in[17];
  const float* bn1_g   = (const float*)d_in[18];
  const float* bn1_b   = (const float*)d_in[19];
  const float* bn1_m   = (const float*)d_in[20];
  const float* bn1_v   = (const float*)d_in[21];
  const float* gam_Wc2 = (const float*)d_in[22];
  const float* gam_bc2 = (const float*)d_in[23];
  const float* bn2_g   = (const float*)d_in[24];
  const float* bn2_b   = (const float*)d_in[25];
  const float* bn2_m   = (const float*)d_in[26];
  const float* bn2_v   = (const float*)d_in[27];

  const int B = 4, C = 512, N = 4096, D1 = 64, Cr = 128;
  const long long CN = (long long)C * N;          // 2,097,152 per batch
  const long long szQ = (long long)B * D1 * N;    // 1,048,576
  const long long szR4 = (long long)B * Cr * N;   // 2,097,152
  const long long szBig = (long long)B * C * N;   // 8,388,608

  // Workspace layout with liveness-based reuse (peak ~200MB fp32).
  float* ws = (float*)d_ws;
  float* R4 = ws;                 // bufQ+bufK -> bufT1 -> bufSp1
  float* R1 = R4 + szR4;          // bufV -> bufCh -> bufSp2 -> bufAtt
  float* R2 = R1 + szBig;         // out1 -> bufX2 -> bufProj
  float* R3 = R2 + szBig;         // bufG
  float* R5 = R3 + szBig;         // bufQKV (B,1536,N)
  float* bufQ = R4;
  float* bufK = R4 + szQ;
  float* bufT1 = R4;
  float* bufSp1 = R4;
  float* bufV = R1;
  float* bufCh = R1;
  float* bufSp2 = R1;
  float* bufAtt = R1;
  float* out1 = R2;
  float* bufX2 = R2;
  float* bufProj = R2;
  float* bufG = R3;
  float* bufQKV = R5;

  dim3 blk(256);
  const long long tot = szBig;

  hipFuncSetAttribute(reinterpret_cast<const void*>(&flash_attn),
                      hipFuncAttributeMaxDynamicSharedMemorySize,
                      (int)flash_smem_bytes(512));

  // 1) q/k/v 1x1 convs as W.X GEMMs
  gemm_wx<0><<<dim3(32, 1, B), blk, 0, stream>>>(Wq, x, bq, bufQ, 64, 512, N,
                                                 CN, (long long)64 * N);
  gemm_wx<0><<<dim3(32, 1, B), blk, 0, stream>>>(Wk, x, bk, bufK, 64, 512, N,
                                                 CN, (long long)64 * N);
  gemm_wx<0><<<dim3(32, 8, B), blk, 0, stream>>>(Wv, x, bv, bufV, 512, 512, N,
                                                 CN, CN);
  // 2) first attention (D=64), epilogue gamma*O + x, out in (C,N) layout
  flash_attn<<<dim3(64, 1, B), blk, flash_smem_bytes(64), stream>>>(
      bufQ, N, (long long)64 * N, bufK, N, (long long)64 * N, bufV, N, CN,
      out1, 1, N, CN, N, 64, 6, 1.0f, gamma, x, CN);
  // 3) GAM channel MLP (per-pixel): t1 = relu(W1.out1+b1); ch = W2.t1+b2
  gemm_wx<1><<<dim3(32, 2, B), blk, 0, stream>>>(gam_W1, out1, gam_b1, bufT1,
                                                 128, 512, N, CN,
                                                 (long long)128 * N);
  gemm_wx<0><<<dim3(32, 8, B), blk, 0, stream>>>(gam_W2, bufT1, gam_b2, bufCh,
                                                 512, 128, N,
                                                 (long long)128 * N, CN);
  ew_mul<<<dim3((unsigned)((tot + 255) / 256)), blk, 0, stream>>>(out1, bufCh,
                                                                  bufG, tot);
  // 4) spatial branch: 7x7 conv + BN (+relu), then 7x7 conv + BN
  conv7_bn<1><<<dim3(64, 2, B), blk, 0, stream>>>(
      gam_Wc1, gam_bc1, bn1_g, bn1_b, bn1_m, bn1_v, bufG, bufSp1, 128, 512, CN,
      (long long)128 * N);
  conv7_bn<0><<<dim3(64, 8, B), blk, 0, stream>>>(
      gam_Wc2, gam_bc2, bn2_g, bn2_b, bn2_m, bn2_v, bufSp1, bufSp2, 512, 128,
      (long long)128 * N, CN);
  // 5) sigmoid spatial gate fused with simam
  simam_gate<<<dim3(B * C), blk, 0, stream>>>(bufG, bufSp2, bufX2);
  // 6) qkv projection, stored transposed (B, 3C, N) so flash reads (d,N) cols
  gemm_wx<0><<<dim3(32, 24, B), blk, 0, stream>>>(W_qkv, bufX2, b_qkv, bufQKV,
                                                  1536, 512, N, CN,
                                                  (long long)1536 * N);
  // 7) second attention (D=512, scale=C^-0.5), out in (C,N) layout
  flash_attn<<<dim3(64, 1, B), blk, flash_smem_bytes(512), stream>>>(
      bufQKV, N, (long long)1536 * N, bufQKV + (long long)512 * N, N,
      (long long)1536 * N, bufQKV + (long long)1024 * N, N,
      (long long)1536 * N, bufAtt, 1, N, CN, N, 512, 9, 0.044194173824159216f,
      nullptr, nullptr, 0);
  // 8) output projection + final permute(0,3,2,1)
  gemm_wx<0><<<dim3(32, 8, B), blk, 0, stream>>>(W_proj, bufAtt, b_proj,
                                                 bufProj, 512, 512, N, CN, CN);
  final_permute<<<dim3((unsigned)(tot / 256)), blk, 0, stream>>>(
      bufProj, (float*)d_out);
}
